// Attention_32143535243924
// MI455X (gfx1250) — matrix operand order repaired and checked by reading, not compiled, not run on hardware
//
#include <hip/hip_runtime.h>
#include <hip/hip_bf16.h>

typedef __attribute__((ext_vector_type(16))) __bf16 v16bf;
typedef __attribute__((ext_vector_type(8)))  __bf16 v8bf;
typedef __attribute__((ext_vector_type(4)))  __bf16 v4bf;
typedef __attribute__((ext_vector_type(8)))  float  v8f;

namespace {
constexpr int kBn = 4, kH = 16, kS = 2048, kD = 128;
constexpr int KTS = 136;   // Kt row stride (bf16): 272B rows, 16B aligned
constexpr int VTS = 72;    // Vt row stride (bf16): 144B rows, 16B aligned
// base-2 softmax domain: fold log2(e) into scale and mask constants
constexpr float kScale = 1.4426950408889634f * 0.08838834764831845f; // log2e/sqrt(128)
constexpr float kNeg2  = -1.442695e9f;                               // -1e9 * log2e
}

__global__ __launch_bounds__(256)
void fa_fwd_gfx1250(const float* __restrict__ Q, const float* __restrict__ K,
                    const float* __restrict__ V, const int* __restrict__ Msk,
                    float* __restrict__ O)
{
    __shared__ __bf16 Kt[64 * KTS];    // K tile, row-major [key][d]
    __shared__ __bf16 Vt[kD * VTS];    // V tile, transposed [d][key]
    __shared__ float  maskAdd[64];     // additive padding mask for this key block

    const int tid  = threadIdx.x;
    const int lane = tid & 31;
    const int wave = tid >> 5;
    const int n    = lane & 15;        // C-layout column (query); A-layout row M
    const int hh   = lane >> 4;        // lane half

    const int bh    = blockIdx.y;      // b*H + h
    const int b     = bh / kH;
    const int qBase = blockIdx.x * 128;
    const int q0    = qBase + wave * 16;
    const int qg    = q0 + n;          // this lane's query row

    const float* Qp = Q + (size_t)bh * kS * kD;
    const float* Kp = K + (size_t)bh * kS * kD;
    const float* Vp = V + (size_t)bh * kS * kD;
    const int*   Mp = Msk + (size_t)b * kS;
    float*       Op = O + (size_t)bh * kS * kD;

    auto wmma = [](v16bf a, v16bf bm, v8f c) -> v8f {
        return __builtin_amdgcn_wmma_f32_16x16x32_bf16(
            false, a, false, bm, (short)0, c, false, false);
    };
    auto cat8 = [](v8bf lo, v8bf hi) -> v16bf {
        return __builtin_shufflevector(lo, hi,
            0,1,2,3,4,5,6,7,8,9,10,11,12,13,14,15);
    };

    // ---- Q^T B-fragments (loop invariant), log2e/sqrt(d) folded in ----
    // B fragment, this code's contiguous k order: lane holds col n, K = 16*hh + j
    v16bf qf[4];
    {
        const float* qrow = Qp + (size_t)qg * kD;
        #pragma unroll
        for (int c = 0; c < 4; ++c) {
            const float4* s4 = (const float4*)(qrow + c * 32 + hh * 16);
            float tmp[16];
            ((float4*)tmp)[0] = s4[0];
            ((float4*)tmp)[1] = s4[1];
            ((float4*)tmp)[2] = s4[2];
            ((float4*)tmp)[3] = s4[3];
            #pragma unroll
            for (int j = 0; j < 16; ++j)
                qf[c][j] = (__bf16)(tmp[j] * kScale);
        }
    }

    // ---- O^T accumulators: 8 tiles of (16 d x 16 q), lane col = query ----
    v8f acc[8];
    #pragma unroll
    for (int t = 0; t < 8; ++t)
        #pragma unroll
        for (int r = 0; r < 8; ++r) acc[t][r] = 0.0f;

    float m_run = -3.0e38f;
    float l_run = 0.0f;

    const int nkb = blockIdx.x * 2 + 2;   // 64-key blocks up to qBase+127

    for (int kb = 0; kb < nkb; ++kb) {
        __syncthreads();

        // ---- cooperative K/V tile load: f32 -> bf16 into LDS ----
        const float4* Kg4 = (const float4*)(Kp + (size_t)kb * 64 * kD);
        const float4* Vg4 = (const float4*)(Vp + (size_t)kb * 64 * kD);
        #pragma unroll
        for (int i = 0; i < 8; ++i) {
            const int idx = tid + 256 * i;     // 2048 float4s = 64x128 floats
            const int row = idx >> 5;          // key within tile
            const int col = (idx & 31) * 4;    // d
            float4 kv = Kg4[idx];
            v4bf k4;
            k4[0] = (__bf16)kv.x; k4[1] = (__bf16)kv.y;
            k4[2] = (__bf16)kv.z; k4[3] = (__bf16)kv.w;
            *(v4bf*)&Kt[row * KTS + col] = k4;
            float4 vv = Vg4[idx];
            Vt[(col + 0) * VTS + row] = (__bf16)vv.x;
            Vt[(col + 1) * VTS + row] = (__bf16)vv.y;
            Vt[(col + 2) * VTS + row] = (__bf16)vv.z;
            Vt[(col + 3) * VTS + row] = (__bf16)vv.w;
        }
        if (tid < 64)
            maskAdd[tid] = (Mp[kb * 64 + tid] == 0) ? kNeg2 : 0.0f;

        // prefetch next K/V tile (global_prefetch_b8): 256*128B = 32KB tile
        if (kb + 1 < nkb) {
            const char* nk = (const char*)(Kp + (size_t)(kb + 1) * 64 * kD);
            const char* nv = (const char*)(Vp + (size_t)(kb + 1) * 64 * kD);
            __builtin_prefetch(nk + tid * 128, 0, 0);
            __builtin_prefetch(nv + tid * 128, 0, 0);
        }

        __syncthreads();

        const int key0 = kb * 64;
        if (key0 > q0 + 15) continue;   // causal: wave done, keep barrier sync

        // ---- S^T = K_tile . Q^T : four 16x16 (key x query) tiles ----
        // A layout: lane row m=n; element j -> d = 32c + 16*hh + j
        v8f ct[4];
        #pragma unroll
        for (int t = 0; t < 4; ++t)
            #pragma unroll
            for (int r = 0; r < 8; ++r) ct[t][r] = 0.0f;
        #pragma unroll
        for (int c = 0; c < 4; ++c) {
            #pragma unroll
            for (int t = 0; t < 4; ++t) {
                const __bf16* kr = &Kt[(t * 16 + n) * KTS + c * 32 + hh * 16];
                v16bf a = cat8(*(const v8bf*)kr, *(const v8bf*)(kr + 8));
                ct[t] = wmma(a, qf[c], ct[t]);
            }
        }

        // ---- masking; lane holds keys t*16 + r + 8*hh ----
        float p[4][8];
        if (key0 + 63 > q0) {          // diagonal block: padding + causal
            #pragma unroll
            for (int t = 0; t < 4; ++t)
                #pragma unroll
                for (int r = 0; r < 8; ++r) {
                    const int kl = t * 16 + r + 8 * hh;
                    p[t][r] = ct[t][r] + maskAdd[kl]
                            + ((key0 + kl > qg) ? kNeg2 : 0.0f);
                }
        } else {                        // strictly past: padding mask only
            #pragma unroll
            for (int t = 0; t < 4; ++t)
                #pragma unroll
                for (int r = 0; r < 8; ++r)
                    p[t][r] = ct[t][r] + maskAdd[t * 16 + r + 8 * hh];
        }

        // ---- online softmax, base-2 domain (lane-local + 1 cross-half shfl) ----
        float bm = p[0][0];
        #pragma unroll
        for (int t = 0; t < 4; ++t)
            #pragma unroll
            for (int r = 0; r < 8; ++r) bm = fmaxf(bm, p[t][r]);
        bm = fmaxf(bm, __shfl_xor(bm, 16, 32));
        const float m_new = fmaxf(m_run, bm);
        const float corr  = exp2f(m_run - m_new);
        float psum = 0.0f;
        #pragma unroll
        for (int t = 0; t < 4; ++t)
            #pragma unroll
            for (int r = 0; r < 8; ++r) {
                p[t][r] = exp2f(p[t][r] - m_new);
                psum += p[t][r];
            }
        psum += __shfl_xor(psum, 16, 32);
        l_run = l_run * corr + psum;
        m_run = m_new;
        #pragma unroll
        for (int t = 0; t < 8; ++t)
            #pragma unroll
            for (int r = 0; r < 8; ++r) acc[t][r] *= corr;

        // ---- P^T C-layout -> two bf16 B-fragments (keys 0..31, 32..63) ----
        v16bf pb0, pb1;
        #pragma unroll
        for (int r = 0; r < 8; ++r) {
            const float xa0 = __shfl_xor(p[0][r], 16, 32);
            const float xa1 = __shfl_xor(p[1][r], 16, 32);
            pb0[r]     = (__bf16)((hh == 0) ? p[0][r] : xa1);
            pb0[r + 8] = (__bf16)((hh == 0) ? xa0 : p[1][r]);
            const float xb0 = __shfl_xor(p[2][r], 16, 32);
            const float xb1 = __shfl_xor(p[3][r], 16, 32);
            pb1[r]     = (__bf16)((hh == 0) ? p[2][r] : xb1);
            pb1[r + 8] = (__bf16)((hh == 0) ? xb0 : p[3][r]);
        }

        // ---- O^T += V^T . P^T : 8 tiles x two K=32 chunks ----
        // A element j reads Vt[16t + n][keyBase + 16*hh + j]
        #pragma unroll
        for (int t = 0; t < 8; ++t) {
            const __bf16* vr = &Vt[(t * 16 + n) * VTS + hh * 16];
            v16bf a0 = cat8(*(const v8bf*)vr,        *(const v8bf*)(vr + 8));
            acc[t] = wmma(a0, pb0, acc[t]);
            v16bf a1 = cat8(*(const v8bf*)(vr + 32), *(const v8bf*)(vr + 40));
            acc[t] = wmma(a1, pb1, acc[t]);
        }
    }

    // ---- epilogue: O = O^T normalized; lane col = query, rows = d ----
    const float inv_l = 1.0f / l_run;
    #pragma unroll
    for (int t = 0; t < 8; ++t) {
        float* orow = Op + (size_t)qg * kD + t * 16 + hh * 8;   // d = 16t + 8hh + r
        float4 o0, o1;
        o0.x = acc[t][0] * inv_l; o0.y = acc[t][1] * inv_l;
        o0.z = acc[t][2] * inv_l; o0.w = acc[t][3] * inv_l;
        o1.x = acc[t][4] * inv_l; o1.y = acc[t][5] * inv_l;
        o1.z = acc[t][6] * inv_l; o1.w = acc[t][7] * inv_l;
        ((float4*)orow)[0] = o0;
        ((float4*)orow)[1] = o1;
    }
}

extern "C" void kernel_launch(void* const* d_in, const int* in_sizes, int n_in,
                              void* d_out, int out_size, void* d_ws, size_t ws_size,
                              hipStream_t stream) {
    (void)in_sizes; (void)n_in; (void)out_size; (void)d_ws; (void)ws_size;
    const float* q = (const float*)d_in[0];
    const float* k = (const float*)d_in[1];
    const float* v = (const float*)d_in[2];
    const int*   m = (const int*)d_in[3];
    float*       o = (float*)d_out;
    dim3 grid(kS / 128, kBn * kH);   // (16, 64): query tiles x (batch*heads)
    fa_fwd_gfx1250<<<grid, 256, 0, stream>>>(q, k, v, m, o);
}
